// SwinTransformerV2Decoder_15272903705107
// MI455X (gfx1250) — compile-verified, matrix operand order checked by reading, not run on hardware
//
#include <hip/hip_runtime.h>
#include <math.h>

typedef unsigned short u16;
typedef __attribute__((ext_vector_type(16))) __bf16 v16bf;
typedef __attribute__((ext_vector_type(8)))  float  v8f;
typedef __attribute__((ext_vector_type(4)))  unsigned int u32x4;

// ---------- bf16 helpers (RNE via bit ops; no __bf16 arithmetic) ----------
__device__ __forceinline__ u16 f2b(float f){
  unsigned u = __builtin_bit_cast(unsigned, f);
  unsigned r = u + 0x7FFFu + ((u >> 16) & 1u);
  return (u16)(r >> 16);
}

// Fragment loader: with row stride Kp (multiple of 32 bf16 = 64B), lane's fragment
// for K-step k0 is two contiguous 16B chunks at chunk indices (k0/8+half) and (+2).
// (ISA 7.12.2: VGPR halves hold K = {0..7,16..23}+8*half relative to k0.)
__device__ __forceinline__ v16bf load_frag(const u32x4* __restrict__ p, int ci){
  u32x4 lo = p[ci];
  u32x4 hi = p[ci + 2];
  return __builtin_bit_cast(v16bf, __builtin_shufflevector(lo, hi, 0,1,2,3,4,5,6,7));
}
__device__ __forceinline__ v8f wmma_bf16(v16bf a, v16bf b, v8f c){
  return __builtin_amdgcn_wmma_f32_16x16x32_bf16(false, a, false, b, (short)0, c, false, false);
}

// ---------------- weight convert + transpose + pad: (K,N) f32 -> (Np,Kp) bf16 ----------------
__global__ void k_cvt_t(const float* __restrict__ in, u16* __restrict__ out,
                        int K, int N, int Kp, int total){
  int t = blockIdx.x*blockDim.x + threadIdx.x;
  if (t >= total) return;                 // total = Np*Kp
  int k = t % Kp, n = t / Kp;
  out[t] = (k < K && n < N) ? f2b(in[(size_t)k*N + n]) : (u16)0;
}

// row-major f32 (rows,C) -> bf16 (rows,Cp) zero-padded
__global__ void k_cvt_pad(const float* __restrict__ in, u16* __restrict__ out,
                          int C, int Cp, int total){
  int t = blockIdx.x*blockDim.x + threadIdx.x;
  if (t >= total) return;
  int c = t % Cp; int row = t / Cp;
  out[t] = (c < C) ? f2b(in[(size_t)row*C + c]) : (u16)0;
}

// zero pad columns [C,Cp) of a bf16 (rows,Cp) buffer
__global__ void k_zpad(u16* __restrict__ buf, int C, int Cp, int total){
  int t = blockIdx.x*blockDim.x + threadIdx.x;
  if (t >= total) return;                 // total = rows*(Cp-C)
  int pad = Cp - C; int c = C + t % pad; int row = t / pad;
  buf[(size_t)row*Cp + c] = 0;
}

// qkv bias = concat(q_b, zeros, v_b)
__global__ void k_qkvb(const float* __restrict__ qb, const float* __restrict__ vbp,
                       float* __restrict__ out, int C){
  int i = blockIdx.x*blockDim.x + threadIdx.x;
  if (i >= 3*C) return;
  out[i] = (i < C) ? qb[i] : (i < 2*C ? 0.f : vbp[i-2*C]);
}

// ---------------- CPB MLP: tab = relu(T @ w1 + b1) @ w2 ----------------
__device__ __forceinline__ float relcoord(int i){
  float r = (float)(i - 7) * (8.0f/7.0f);
  float s = (r > 0.f) ? 1.f : (r < 0.f ? -1.f : 0.f);
  return s * log2f(fabsf(r) + 1.f) * (1.f/3.0f);     // /log2(8)
}
__global__ void k_cpb_tab(const float* __restrict__ w1, const float* __restrict__ b1,
                          const float* __restrict__ w2, float* __restrict__ tab, int h){
  int t = blockIdx.x*blockDim.x + threadIdx.x;
  if (t >= 225*h) return;
  int head = t % h, idx = t / h;
  float T0 = relcoord(idx/15), T1 = relcoord(idx%15);
  float acc = 0.f;
  for (int u = 0; u < 512; ++u){
    float hu = T0*w1[u] + T1*w1[512+u] + b1[u];
    hu = hu > 0.f ? hu : 0.f;
    acc += hu * w2[u*h + head];
  }
  tab[idx*h + head] = acc;
}
__global__ void k_cpb_bias(const float* __restrict__ tab, float* __restrict__ bias, int h){
  int t = blockIdx.x*blockDim.x + threadIdx.x;
  if (t >= h*4096) return;
  int m = t & 63, n = (t >> 6) & 63, head = t >> 12;
  int d0 = (n>>3) - (m>>3) + 7, d1 = (n&7) - (m&7) + 7;
  float v = tab[(d0*15 + d1)*h + head];
  bias[t] = 16.f / (1.f + expf(-v));
}

// ---------------- roll(-shift) + window_partition -> bf16 (M,Cp) padded ----------------
__global__ void k_winpart(const float* __restrict__ x, u16* __restrict__ xw,
                          int H, int W, int C, int Cp, int shift, int total){
  int t = blockIdx.x*blockDim.x + threadIdx.x;
  if (t >= total) return;
  int c = t % Cp; int rest = t / Cp; int n = rest & 63; int bw = rest >> 6;
  if (c >= C){ xw[t] = 0; return; }
  int nWx = W >> 3, nW = (H >> 3)*nWx;
  int b = bw / nW, w = bw % nW;
  int yy = (w / nWx)*8 + (n >> 3);
  int xx = (w % nWx)*8 + (n & 7);
  int y  = yy + shift; if (y  >= H) y  -= H;   // rolled[yy] = orig[(yy+s) mod H]
  int xp = xx + shift; if (xp >= W) xp -= W;
  xw[t] = f2b(x[((size_t)(b*H + y)*W + xp)*C + c]);
}

// ---------------- generic bf16 WMMA GEMM, 16x64 per wave (A-fragment reuse x4) ----------------
// A: (M,Kp) row-major, Bt: (roundup(N,64),Kp) transposed+padded. Fused bias/GELU/bf16 store.
__global__ void k_gemm(const u16* __restrict__ A, const u16* __restrict__ Bt,
                       const float* __restrict__ bias,
                       float* __restrict__ outF, u16* __restrict__ outB,
                       int M, int N, int Kp, int act){
  int lane  = threadIdx.x & 31;
  int tileM = (blockIdx.x*blockDim.y + threadIdx.y) << 4;
  int tileN = blockIdx.y << 6;                       // 4 N-tiles per wave
  if (tileM >= M) return;
  int half = lane >> 4, lid = lane & 15;
  int m0 = tileM + lid;
  const u32x4* Ar = (const u32x4*)(A + (size_t)m0*Kp);
  const u32x4* Br[4];
#pragma unroll
  for (int j = 0; j < 4; ++j)
    Br[j] = (const u32x4*)(Bt + (size_t)(tileN + j*16 + lid)*Kp);
  v8f acc[4] = {{},{},{},{}};
  for (int k0 = 0; k0 < Kp; k0 += 32){
    int ci = (k0 >> 3) + half;
    v16bf a  = load_frag(Ar, ci);
    v16bf b0 = load_frag(Br[0], ci);
    v16bf b1 = load_frag(Br[1], ci);
    v16bf b2 = load_frag(Br[2], ci);
    v16bf b3 = load_frag(Br[3], ci);
    acc[0] = wmma_bf16(a, b0, acc[0]);
    acc[1] = wmma_bf16(a, b1, acc[1]);
    acc[2] = wmma_bf16(a, b2, acc[2]);
    acc[3] = wmma_bf16(a, b3, acc[3]);
  }
#pragma unroll
  for (int j = 0; j < 4; ++j){
    int n0 = tileN + j*16 + lid;
    if (n0 >= N) continue;
    float bv = bias ? bias[n0] : 0.f;
#pragma unroll
    for (int r = 0; r < 8; ++r){
      float v = acc[j][r] + bv;
      if (act == 1) v = 0.5f*v*(1.f + erff(v*0.70710678118f));   // exact GELU
      size_t o = (size_t)(tileM + r + (half<<3))*N + n0;
      if (outF) outF[o] = v; else outB[o] = f2b(v);
    }
  }
}

// ---------------- qkv post: split heads, cosine-normalize q,k; v stored transposed ----------------
// qn,kn: (bh, n, hdp) zero-padded; vt: (bh, hdp, 64) zero-padded rows
__global__ void k_qkv_post(const float* __restrict__ qkv, u16* __restrict__ qn,
                           u16* __restrict__ kn, u16* __restrict__ vt,
                           int heads, int hd, int hdp, int C, int total){
  int t = blockIdx.x*blockDim.x + threadIdx.x;
  if (t >= total) return;                          // t = (b_*heads+h)*64 + n
  int n = t & 63; int bh = t >> 6;
  int b_ = bh / heads, h = bh % heads;
  const float* base = qkv + ((size_t)(b_*64 + n))*3*C + h*hd;
  float sq = 0.f, sk = 0.f;
  for (int d = 0; d < hd; ++d){
    float q = base[d], k = base[C + d];
    sq += q*q; sk += k*k;
  }
  float rq = 1.f / fmaxf(sqrtf(sq), 1e-12f);
  float rk = 1.f / fmaxf(sqrtf(sk), 1e-12f);
  size_t o = (size_t)t * hdp;
  for (int d = 0; d < hd; ++d){
    qn[o+d] = f2b(base[d]*rq);
    kn[o+d] = f2b(base[C+d]*rk);
    vt[((size_t)bh*hdp + d)*64 + n] = f2b(base[2*C+d]);
  }
  for (int d = hd; d < hdp; ++d){
    qn[o+d] = 0; kn[o+d] = 0;
    vt[((size_t)bh*hdp + d)*64 + n] = 0;
  }
}

// ---------------- batched scores: 16x64 strip per wave, A reused across 4 wmmas ----------------
__global__ void k_scores(const u16* __restrict__ qn, const u16* __restrict__ kn,
                         float* __restrict__ attn, int hdp){
  int lane = threadIdx.x & 31;
  int bh = blockIdx.y;
  int tm = blockIdx.x << 4;                         // grid.x = 4
  int half = lane >> 4, lid = lane & 15;
  const u32x4* Q = (const u32x4*)(qn + ((size_t)bh*64 + tm + lid)*hdp);
  const u32x4* Kb[4];
#pragma unroll
  for (int j = 0; j < 4; ++j)
    Kb[j] = (const u32x4*)(kn + ((size_t)bh*64 + j*16 + lid)*hdp);
  v8f acc[4] = {{},{},{},{}};
  for (int k0 = 0; k0 < hdp; k0 += 32){
    int ci = (k0 >> 3) + half;
    v16bf a = load_frag(Q, ci);
    acc[0] = wmma_bf16(a, load_frag(Kb[0], ci), acc[0]);
    acc[1] = wmma_bf16(a, load_frag(Kb[1], ci), acc[1]);
    acc[2] = wmma_bf16(a, load_frag(Kb[2], ci), acc[2]);
    acc[3] = wmma_bf16(a, load_frag(Kb[3], ci), acc[3]);
  }
  float* o = attn + (size_t)bh*4096 + lid;
#pragma unroll
  for (int j = 0; j < 4; ++j)
#pragma unroll
    for (int r = 0; r < 8; ++r)
      o[(tm + r + (half<<3))*64 + j*16] = acc[j][r];
}

// ---------------- scale + CPB bias + analytic shifted-window mask + softmax -> bf16 ----------------
__global__ void k_softmax(const float* __restrict__ attn, u16* __restrict__ ap,
                          const float* __restrict__ bias, const float* __restrict__ ls,
                          int heads, int nW, int nWx, int H, int W, int shift){
  int row = blockIdx.x;                 // (b_*heads + h)*64 + n
  int m = threadIdx.x;
  int n = row & 63; int bh = row >> 6;
  int h = bh % heads; int b_ = bh / heads;
  float scale = expf(fminf(ls[h], 4.60517018599f));   // exp(min(log_scale, ln 100))
  float v = attn[(size_t)row*64 + m]*scale + bias[((size_t)h*64 + n)*64 + m];
  if (shift > 0){
    int w = b_ % nW;
    int wy = (w / nWx) << 3, wx = (w % nWx) << 3;
    int yn = wy + (n>>3), xn_ = wx + (n&7);
    int ym = wy + (m>>3), xm  = wx + (m&7);
    int rn = (yn < H-8 ? 0 : (yn < H-4 ? 1 : 2))*3 + (xn_ < W-8 ? 0 : (xn_ < W-4 ? 1 : 2));
    int rm = (ym < H-8 ? 0 : (ym < H-4 ? 1 : 2))*3 + (xm  < W-8 ? 0 : (xm  < W-4 ? 1 : 2));
    if (rn != rm) v -= 100.f;
  }
  __shared__ float sh[64];
  sh[m] = v; __syncthreads();
  for (int st = 32; st > 0; st >>= 1){ if (m < st) sh[m] = fmaxf(sh[m], sh[m+st]); __syncthreads(); }
  float mx = sh[0]; __syncthreads();
  float e = expf(v - mx);
  sh[m] = e; __syncthreads();
  for (int st = 32; st > 0; st >>= 1){ if (m < st) sh[m] += sh[m+st]; __syncthreads(); }
  ap[(size_t)row*64 + m] = f2b(e / sh[0]);
}

// ---------------- batched attn @ v: 16x32 per wave (A reused x2), head-merge store ----------------
__global__ void k_ctx(const u16* __restrict__ ap, const u16* __restrict__ vt,
                      u16* __restrict__ ctx, int heads, int hd, int hdp, int Cp){
  int lane = threadIdx.x & 31;
  int bh = blockIdx.y;
  int tm = blockIdx.x << 4;                         // grid.x = 4
  int b_ = bh / heads, h = bh % heads;
  int half = lane >> 4, lid = lane & 15;
  const u32x4* Ar = (const u32x4*)(ap + ((size_t)bh*64 + tm + lid)*64);
  const u32x4* Br0 = (const u32x4*)(vt + ((size_t)bh*hdp + lid)*64);
  const u32x4* Br1 = (const u32x4*)(vt + ((size_t)bh*hdp + 16 + lid)*64);
  v8f acc[2] = {{},{}};
  for (int k0 = 0; k0 < 64; k0 += 32){
    int ci = (k0 >> 3) + half;
    v16bf a = load_frag(Ar, ci);
    acc[0] = wmma_bf16(a, load_frag(Br0, ci), acc[0]);
    acc[1] = wmma_bf16(a, load_frag(Br1, ci), acc[1]);
  }
#pragma unroll
  for (int j = 0; j < 2; ++j){
    int n0 = j*16 + lid;
    if (n0 >= hd) continue;
#pragma unroll
    for (int r = 0; r < 8; ++r)
      ctx[((size_t)(b_*64 + tm + r + (half<<3)))*Cp + h*hd + n0] = f2b(acc[j][r]);
  }
}

// ---------------- residual/LN. add=1: x += LN(src); add=0: x = LN(src) ----------------
__global__ void k_add_ln(float* __restrict__ x, const float* __restrict__ src,
                         const float* __restrict__ g, const float* __restrict__ b,
                         int C, int add){
  int row = blockIdx.x;
  const float* s = src + (size_t)row*C;
  float* xr = x + (size_t)row*C;
  __shared__ float red[256];
  float acc = 0.f;
  for (int c = threadIdx.x; c < C; c += 256) acc += s[c];
  red[threadIdx.x] = acc; __syncthreads();
  for (int st = 128; st > 0; st >>= 1){ if (threadIdx.x < st) red[threadIdx.x] += red[threadIdx.x+st]; __syncthreads(); }
  float mean = red[0] / C; __syncthreads();
  float vs = 0.f;
  for (int c = threadIdx.x; c < C; c += 256){ float d = s[c]-mean; vs += d*d; }
  red[threadIdx.x] = vs; __syncthreads();
  for (int st = 128; st > 0; st >>= 1){ if (threadIdx.x < st) red[threadIdx.x] += red[threadIdx.x+st]; __syncthreads(); }
  float rstd = rsqrtf(red[0]/C + 1e-5f);
  for (int c = threadIdx.x; c < C; c += 256){
    float v = (s[c]-mean)*rstd*g[c] + b[c];
    xr[c] = add ? (xr[c] + v) : v;
  }
}

// window_reverse + roll(+shift) + residual + post-LN, fused gather
__global__ void k_add_ln_win(float* __restrict__ x, const float* __restrict__ src,
                             const float* __restrict__ g, const float* __restrict__ bb,
                             int H, int W, int C, int shift){
  int t = blockIdx.x;                   // (b, y, x) row-major = token row of x
  int xc = t % W; int rest = t / W; int y = rest % H; int b = rest / H;
  int yy = y  - shift; if (yy < 0) yy += H;
  int xx = xc - shift; if (xx < 0) xx += W;
  int nWx = W >> 3, nW = (H >> 3)*nWx;
  int bw = b*nW + (yy>>3)*nWx + (xx>>3);
  int n  = ((yy & 7) << 3) | (xx & 7);
  const float* s = src + ((size_t)(bw*64 + n))*C;
  float* xr = x + (size_t)t * C;
  __shared__ float red[256];
  float acc = 0.f;
  for (int c = threadIdx.x; c < C; c += 256) acc += s[c];
  red[threadIdx.x] = acc; __syncthreads();
  for (int st = 128; st > 0; st >>= 1){ if (threadIdx.x < st) red[threadIdx.x] += red[threadIdx.x+st]; __syncthreads(); }
  float mean = red[0] / C; __syncthreads();
  float vs = 0.f;
  for (int c = threadIdx.x; c < C; c += 256){ float d = s[c]-mean; vs += d*d; }
  red[threadIdx.x] = vs; __syncthreads();
  for (int st = 128; st > 0; st >>= 1){ if (threadIdx.x < st) red[threadIdx.x] += red[threadIdx.x+st]; __syncthreads(); }
  float rstd = rsqrtf(red[0]/C + 1e-5f);
  for (int c = threadIdx.x; c < C; c += 256)
    xr[c] += (s[c]-mean)*rstd*g[c] + bb[c];
}

// ---------------- PatchSplit gather: (B,H,W,C) -> (B,2H,2W,C4p) bf16 padded ----------------
__global__ void k_patch_gather(const float* __restrict__ x, u16* __restrict__ xs,
                               int H, int W, int C, int C4p, int total){
  int t = blockIdx.x*blockDim.x + threadIdx.x;
  if (t >= total) return;
  int C4 = C >> 2;
  int c = t % C4p; int rest = t / C4p;
  if (c >= C4){ xs[t] = 0; return; }
  int xq = rest % (2*W); rest /= (2*W);
  int yq = rest % (2*H); int b = rest / (2*H);
  int hh = yq >> 1, a = yq & 1, ww = xq >> 1, b2 = xq & 1;
  int pidx = (a << 1) | b2;
  int grp = (pidx == 1) ? 2 : (pidx == 2 ? 1 : pidx);   // perm [0,2,1,3]
  xs[t] = f2b(x[((size_t)(b*H + hh)*W + ww)*C + grp*C4 + c]);
}

// ---------------- final 3x3 conv (Cin=24, Cout=3, pad 1), NHWC in -> NCHW out ----------------
__global__ void k_conv3x3(const float* __restrict__ xn, const float* __restrict__ w,
                          const float* __restrict__ cb, float* __restrict__ out,
                          int H, int W){
  int t = blockIdx.x*blockDim.x + threadIdx.x;
  int total = 2*3*H*W;
  if (t >= total) return;
  int x = t % W; int rest = t / W; int y = rest % H; rest /= H;
  int o = rest % 3; int b = rest / 3;
  float acc = cb[o];
  for (int ky = 0; ky < 3; ++ky){
    int yy = y + ky - 1; if (yy < 0 || yy >= H) continue;
    for (int kx = 0; kx < 3; ++kx){
      int xx = x + kx - 1; if (xx < 0 || xx >= W) continue;
      const float* p  = xn + ((size_t)(b*H + yy)*W + xx)*24;
      const float* wp = w + (size_t)o*216 + ky*3 + kx;    // w[o][c][ky][kx]
      for (int c = 0; c < 24; ++c) acc += p[c] * wp[c*9];
    }
  }
  out[t] = acc;
}

// ======================= host orchestration =======================
static const int DIMS_H[5]  = {768, 384, 192, 96, 48};
static const int HEADS_H[5] = {24, 12, 6, 3, 2};
static inline int rnd(int v, int a){ return (v + a - 1) & ~(a - 1); }

extern "C" void kernel_launch(void* const* d_in, const int* in_sizes, int n_in,
                              void* d_out, int out_size, void* d_ws, size_t ws_size,
                              hipStream_t stream) {
  (void)in_sizes; (void)n_in; (void)out_size; (void)ws_size;
  auto P = [&](int i)->const float* { return (const float*)d_in[i]; };
  auto cdiv = [](long a, long b)->unsigned { return (unsigned)((a + b - 1) / b); };

  // workspace arena
  char* base = (char*)d_ws;
  size_t off = 0;
  auto carve = [&](size_t bytes)->void* {
    void* p = base + off; off = (off + bytes + 255) & ~(size_t)255; return p;
  };
  const size_t MAXX = 12582912;                 // 2*512*512*24 (largest x)
  float* xA   = (float*)carve(MAXX*4);
  float* xB   = (float*)carve(MAXX*4);
  u16*   xbf  = (u16*)  carve(16777216ull*2);   // padded A staging (win-part/MLP/gather)
  float* qkv  = (float*)carve(18874368ull*4);   // B*L*3C max; reused as h1 (bf16)
  u16*   h1   = (u16*)qkv;
  u16*   qn   = (u16*)  carve(8388608ull*2);    // (bh,64,hdp)
  u16*   kn   = (u16*)  carve(8388608ull*2);
  u16*   vt   = (u16*)  carve(8388608ull*2);    // (bh,hdp,64)
  float* attn = (float*)carve(16777216ull*4);   // reused as h2 (fp32)
  float* h2   = attn;
  u16*   attp = (u16*)  carve(16777216ull*2);
  u16*   ctx  = (u16*)  carve(8388608ull*2);    // (M,Cp)
  float* aout = (float*)carve(6291456ull*4);    // (M,C)
  u16*   wbuf = (u16*)  carve(2359296ull*2);    // largest weight Bt (Np64,Kp)
  float* cbias= (float*)carve((size_t)24*64*64*4);
  float* tab  = (float*)carve((size_t)225*24*4);
  float* qkvb = (float*)carve(2304*4);

  auto gemm = [&](const u16* A, const u16* Bt, const float* bias,
                  float* oF, u16* oB, int M, int N, int Kp, int act){
    dim3 grid(cdiv(M, 64), cdiv(N, 64));      // 4 waves x (16x64) per block
    dim3 blk(32, 4);
    k_gemm<<<grid, blk, 0, stream>>>(A, Bt, bias, oF, oB, M, N, Kp, act);
  };
  auto cvtw = [&](const float* w, int K, int N, int Kp){
    int Np = rnd(N, 64);                      // pad Bt rows to 4 N-tiles
    k_cvt_t<<<cdiv((long)Np*Kp,256),256,0,stream>>>(w, wbuf, K, N, Kp, Np*Kp);
  };

  hipMemcpyAsync(xA, d_in[0], (size_t)2*256*768*sizeof(float),
                 hipMemcpyDeviceToDevice, stream);

  float* xc = xA; float* xalt = xB;
  int H = 16, W = 16;
  for (int li = 0; li < 5; ++li) {
    const int C = DIMS_H[li], heads = HEADS_H[li], hd = C/heads;
    const int Cp = rnd(C, 32), hdp = rnd(hd, 32);
    const int L = H*W, nWx = W/8, nW = (H/8)*nWx, Bw = 2*nW;
    const int M = 2*L;                                    // B * L tokens

    for (int bj = 0; bj < 2; ++bj) {
      const int pb = 1 + li*37 + bj*17;
      const int shift = (bj & 1) ? 4 : 0;

      // qkv bias + CPB relative-position bias
      k_qkvb<<<cdiv(3*C,256),256,0,stream>>>(P(pb+1), P(pb+2), qkvb, C);
      k_cpb_tab<<<cdiv(225*heads,128),128,0,stream>>>(P(pb+4), P(pb+5), P(pb+6), tab, heads);
      k_cpb_bias<<<cdiv(heads*4096,256),256,0,stream>>>(tab, cbias, heads);

      // roll + window partition (padded bf16) ; qkv GEMM
      k_winpart<<<cdiv((long)M*Cp,256),256,0,stream>>>(xc, xbf, H, W, C, Cp, shift, M*Cp);
      cvtw(P(pb+0), C, 3*C, Cp);
      gemm(xbf, wbuf, qkvb, qkv, nullptr, M, 3*C, Cp, 0);

      // split heads + cosine normalize (+ v transposed)
      k_qkv_post<<<cdiv((long)Bw*heads*64,256),256,0,stream>>>(qkv, qn, kn, vt,
                                                heads, hd, hdp, C, Bw*heads*64);
      // scores, softmax(+bias+mask), context
      k_scores<<<dim3(4, (unsigned)(Bw*heads)), 32, 0, stream>>>(qn, kn, attn, hdp);
      k_softmax<<<dim3((unsigned)(Bw*heads*64)), 64, 0, stream>>>(attn, attp, cbias, P(pb+3),
                                                                  heads, nW, nWx, H, W, shift);
      k_ctx<<<dim3(4, (unsigned)(Bw*heads)), 32, 0, stream>>>(attp, vt, ctx,
                                                              heads, hd, hdp, Cp);
      if (Cp > C)
        k_zpad<<<cdiv((long)M*(Cp-C),256),256,0,stream>>>(ctx, C, Cp, M*(Cp-C));
      // proj + (window-reverse + roll + residual + post-LN)
      cvtw(P(pb+7), C, C, Cp);
      gemm(ctx, wbuf, P(pb+8), aout, nullptr, M, C, Cp, 0);
      k_add_ln_win<<<dim3((unsigned)M),256,0,stream>>>(xc, aout, P(pb+9), P(pb+10), H, W, C, shift);

      // MLP: fc1 (GELU, bf16 out) -> fc2 -> residual post-LN
      k_cvt_pad<<<cdiv((long)M*Cp,256),256,0,stream>>>(xc, xbf, C, Cp, M*Cp);
      cvtw(P(pb+11), C, 4*C, Cp);
      gemm(xbf, wbuf, P(pb+12), nullptr, h1, M, 4*C, Cp, 1);
      cvtw(P(pb+13), 4*C, C, 4*C);
      gemm(h1, wbuf, P(pb+14), h2, nullptr, M, C, 4*C, 0);
      k_add_ln<<<dim3((unsigned)M),256,0,stream>>>(xc, h2, P(pb+15), P(pb+16), C, 1);
    }

    // PatchSplit: gather (padded) -> GEMM (C/4 -> C/2) -> LN; 2x2 upsample
    const int ub = 1 + li*37 + 34;
    const int C4p = rnd(C/4, 32);
    k_patch_gather<<<cdiv((long)4*M*C4p,256),256,0,stream>>>(xc, xbf, H, W, C, C4p, 4*M*C4p);
    cvtw(P(ub), C/4, C/2, C4p);
    gemm(xbf, wbuf, nullptr, xalt, nullptr, 4*M, C/2, C4p, 0);
    k_add_ln<<<dim3((unsigned)(4*M)),256,0,stream>>>(xalt, xalt, P(ub+1), P(ub+2), C/2, 0);
    { float* t = xc; xc = xalt; xalt = t; }
    H *= 2; W *= 2;
  }

  // final LN (C=24) into free buffer, then 3x3 erase conv -> d_out (2,3,512,512)
  k_add_ln<<<dim3((unsigned)(2*512*512)),256,0,stream>>>(xalt, xc, P(186), P(187), 24, 0);
  k_conv3x3<<<cdiv((long)2*3*512*512,256),256,0,stream>>>(xalt, P(188), P(189),
                                                          (float*)d_out, 512, 512);
}
